// Attention_63127429316818
// MI455X (gfx1250) — matrix operand order repaired and checked by reading, not compiled, not run on hardware
//
#include <hip/hip_runtime.h>

#define DIMC 512
#define ADIM 256
#define NHEAD 8
#define HD 32
#define NPOS 4096
#define ATT_SCALE_LOG2 0.06377590046818906f  // 512^-0.5 * log2(e)

typedef __attribute__((ext_vector_type(16))) _Float16 v16h;
typedef __attribute__((ext_vector_type(8)))  float    v8f;
typedef int v4i __attribute__((vector_size(16)));

#ifndef __has_builtin
#define __has_builtin(x) 0
#endif

// ---- gfx1250 async global->LDS copy (ASYNCcnt path) --------------------------------
typedef __attribute__((address_space(1))) v4i* gvec4p;
typedef __attribute__((address_space(3))) v4i* lvec4p;

#if __has_builtin(__builtin_amdgcn_global_load_async_to_lds_b128)
__device__ __forceinline__ void async_b128(const void* g, void* l) {
  __builtin_amdgcn_global_load_async_to_lds_b128((gvec4p)g, (lvec4p)l, 0, 0);
}
#else
__device__ __forceinline__ void async_b128(const void* g, void* l) {
  asm volatile("global_load_async_to_lds_b128 %0, %1, off"
               :: "v"((unsigned)(unsigned long long)l),
                  "v"((unsigned long long)g)
               : "memory");
}
#endif

#if __has_builtin(__builtin_amdgcn_s_wait_asynccnt)
#define WAIT_ASYNC0() __builtin_amdgcn_s_wait_asynccnt(0)
#else
#define WAIT_ASYNC0() asm volatile("s_wait_asynccnt 0x0" ::: "memory")
#endif

union FragH { v16h h; unsigned int u[8]; };
union Pack8 { uint4 u4; _Float16 h[8]; };

__device__ __forceinline__ v8f wmma_f16(v16h a, v16h b, v8f c) {
  return __builtin_amdgcn_wmma_f32_16x16x32_f16(false, a, false, b, (short)0, c,
                                                false, false);
}

// A-matrix fragment (16x32 MxK, f16) from a row-major [rows][32] f16 region.
__device__ __forceinline__ v16h load_a_frag(const _Float16* base, int row) {
  const int lane = threadIdx.x & 31;
  const int hf = lane >> 4;
  const unsigned int* rw = (const unsigned int*)(base + (size_t)(row + (lane & 15)) * 32);
  FragH f;
#pragma unroll
  for (int j = 0; j < 8; ++j) f.u[j] = rw[((j & 4) << 1) + hf * 4 + (j & 3)];
  return f.h;
}

// B-matrix fragment (32x16 KxN, f16) from row-major [n][32] f16 region (B^T view).
__device__ __forceinline__ v16h load_b_frag(const _Float16* base, int row) {
  const int lane = threadIdx.x & 31;
  const int hf = lane >> 4;
  const unsigned int* rw = (const unsigned int*)(base + (size_t)(row + (lane & 15)) * 32);
  FragH f;
#pragma unroll
  for (int j = 0; j < 8; ++j) f.u[j] = rw[((j & 4) << 1) + hf * 4 + (j & 3)];
  return f.h;
}

__device__ __forceinline__ void cvt8_store(_Float16* dst, float4 a0, float4 a1) {
  Pack8 pk;
  pk.h[0] = (_Float16)a0.x; pk.h[1] = (_Float16)a0.y;
  pk.h[2] = (_Float16)a0.z; pk.h[3] = (_Float16)a0.w;
  pk.h[4] = (_Float16)a1.x; pk.h[5] = (_Float16)a1.y;
  pk.h[6] = (_Float16)a1.z; pk.h[7] = (_Float16)a1.w;
  *(uint4*)dst = pk.u4;
}

// ---------------- Kernel 1: QKV projection (f32 -> f16) -----------------------------
// q,k stored transposed [bh][p][d]; v stored standard [bh][d][p].
// Double-buffered LDS, one barrier per K-step; staging for step i+1 overlaps WMMA of i.
__global__ __launch_bounds__(256) void cdna5_proj_qkv(
    const float* __restrict__ query, const float* __restrict__ context,
    const float* __restrict__ Wq, const float* __restrict__ Wk,
    const float* __restrict__ Wv,
    _Float16* __restrict__ qT, _Float16* __restrict__ kT, _Float16* __restrict__ vS) {
  const int which = blockIdx.z % 3;
  const int b = blockIdx.z / 3;
  const float* W = (which == 0) ? Wq : (which == 1) ? Wk : Wv;
  const float* X = ((which == 0) ? query : context) + (size_t)b * DIMC * NPOS;
  _Float16* out = (which == 0) ? qT : (which == 1) ? kT : vS;

  const int o0 = blockIdx.y * 64;
  const int p0 = blockIdx.x * 128;
  __shared__ __align__(16) _Float16 lds_a[2][64 * 32];   // [m][kk]
  __shared__ __align__(16) _Float16 lds_b[2][128 * 32];  // [n][kk]
  const int t = threadIdx.x;
  const int lane = t & 31, wv = t >> 5;
  const int wr = wv >> 2, wc = wv & 3;
  v8f acc[2][2] = {};

  // per-thread staging bases (strength-reduced; all loop loads use imm offsets)
  const int am = t >> 2, ac = (t & 3) * 8;    // A: row am, cols ac..ac+7
  const int bk = t >> 3, bn = (t & 7) * 16;   // B: row kk=bk, cols bn..bn+15
  const float* wp = W + (size_t)(o0 + am) * DIMC + ac;
  const float* xp = X + (size_t)bk * NPOS + p0 + bn;

  auto stage = [&](int nb) {
    float4 a0 = *(const float4*)(wp);
    float4 a1 = *(const float4*)(wp + 4);
    cvt8_store(&lds_a[nb][am * 32 + ac], a0, a1);
    float4 x0 = *(const float4*)(xp);
    float4 x1 = *(const float4*)(xp + 4);
    float4 x2 = *(const float4*)(xp + 8);
    float4 x3 = *(const float4*)(xp + 12);
    _Float16* dst = &lds_b[nb][bn * 32 + bk];   // [n][kk]: stride 32 across n
    dst[0 * 32]  = (_Float16)x0.x; dst[1 * 32]  = (_Float16)x0.y;
    dst[2 * 32]  = (_Float16)x0.z; dst[3 * 32]  = (_Float16)x0.w;
    dst[4 * 32]  = (_Float16)x1.x; dst[5 * 32]  = (_Float16)x1.y;
    dst[6 * 32]  = (_Float16)x1.z; dst[7 * 32]  = (_Float16)x1.w;
    dst[8 * 32]  = (_Float16)x2.x; dst[9 * 32]  = (_Float16)x2.y;
    dst[10 * 32] = (_Float16)x2.z; dst[11 * 32] = (_Float16)x2.w;
    dst[12 * 32] = (_Float16)x3.x; dst[13 * 32] = (_Float16)x3.y;
    dst[14 * 32] = (_Float16)x3.z; dst[15 * 32] = (_Float16)x3.w;
    __builtin_prefetch(wp + 64, 0, 1);
    __builtin_prefetch(xp + (size_t)64 * NPOS, 0, 1);
    wp += 32;
    xp += (size_t)32 * NPOS;
  };

  stage(0);
  for (int it = 0; it < 16; ++it) {
    __syncthreads();
    if (it + 1 < 16) stage((it + 1) & 1);
    const _Float16* la = lds_a[it & 1];
    const _Float16* lb = lds_b[it & 1];
    v16h af0 = load_a_frag(la, wr * 32);
    v16h af1 = load_a_frag(la, wr * 32 + 16);
    v16h bf0 = load_b_frag(lb, wc * 32);
    v16h bf1 = load_b_frag(lb, wc * 32 + 16);
    acc[0][0] = wmma_f16(af0, bf0, acc[0][0]);
    acc[0][1] = wmma_f16(af0, bf1, acc[0][1]);
    acc[1][0] = wmma_f16(af1, bf0, acc[1][0]);
    acc[1][1] = wmma_f16(af1, bf1, acc[1][1]);
  }

  const int hf = lane >> 4, n16 = lane & 15;
  const int h = (o0 + wr * 32) >> 5;     // tile never crosses a head
  if (which != 2) {                      // q,k: transposed [bh][p][d], 16B packed store
#pragma unroll
    for (int ti = 0; ti < 2; ++ti)
#pragma unroll
      for (int tj = 0; tj < 2; ++tj) {
        const int d0 = ti * 16 + hf * 8;
        const int p = p0 + wc * 32 + tj * 16 + n16;
        Pack8 pk;
#pragma unroll
        for (int r = 0; r < 8; ++r) pk.h[r] = (_Float16)acc[ti][tj][r];
        *(uint4*)(out + ((size_t)(b * NHEAD + h) * NPOS + p) * HD + d0) = pk.u4;
      }
  } else {                               // v: standard [bh][d][p]
#pragma unroll
    for (int ti = 0; ti < 2; ++ti)
#pragma unroll
      for (int tj = 0; tj < 2; ++tj) {
        const int p = p0 + wc * 32 + tj * 16 + n16;
#pragma unroll
        for (int r = 0; r < 8; ++r) {
          int d = ti * 16 + hf * 8 + r;
          out[((size_t)(b * NHEAD + h) * HD + d) * NPOS + p] = (_Float16)acc[ti][tj][r];
        }
      }
  }
}

// ---------------- Kernel 2: flash attention per (b,h) -------------------------------
// Double-buffered async k/v staging; one barrier per key block; async load of block
// i+1 issued right after the barrier, overlapping WMMA + softmax of block i.
__global__ __launch_bounds__(256) void cdna5_flash_attn(
    const _Float16* __restrict__ qT, const _Float16* __restrict__ kT,
    const _Float16* __restrict__ vS, _Float16* __restrict__ oT) {
  const int bh = blockIdx.y;
  const int b = bh >> 3, h = bh & 7;
  const _Float16* qbh = qT + (size_t)bh * NPOS * HD;
  const _Float16* kbh = kT + (size_t)bh * NPOS * HD;
  const _Float16* vbh = vS + (size_t)bh * HD * NPOS;   // [d][p]
  const int t = threadIdx.x, lane = t & 31, wv = t >> 5;
  const int hf = lane >> 4, n16 = lane & 15;
  const int i0 = blockIdx.x * 128 + wv * 16;

  __shared__ __align__(16) _Float16 lds_k[2][32 * 32];  // [j][d]
  __shared__ __align__(16) _Float16 lds_v[2][32 * 32];  // [d][j]
  __shared__ __align__(16) _Float16 lds_p[8][16 * 32];  // per-wave P tile

  FragH qa;  // q A-fragment held for the whole kernel (K = head_dim = 32)
  {
    const unsigned int* qrow = (const unsigned int*)(qbh + (size_t)(i0 + n16) * HD);
#pragma unroll
    for (int j = 0; j < 8; ++j) qa.u[j] = qrow[((j & 4) << 1) + hf * 4 + (j & 3)];
  }
  float m_i[8], l_i[8];
#pragma unroll
  for (int r = 0; r < 8; ++r) { m_i[r] = -1e30f; l_i[r] = 0.0f; }
  v8f acc0 = {}, acc1 = {};

  // one async B128 per lane per block: waves 0-3 stage k, waves 4-7 stage v
  const int c = t & 127;
  const _Float16* g;
  size_t gstep;
  char *lb0, *lb1;
  if (t < 128) {
    g = kbh + c * 8;                          // [j][d] rows, contiguous
    gstep = (size_t)32 * HD;                  // next 32 keys
    lb0 = (char*)lds_k[0] + c * 16;
    lb1 = (char*)lds_k[1] + c * 16;
  } else {
    g = vbh + (size_t)(c >> 2) * NPOS + (c & 3) * 8;  // row d, 32 cols per block
    gstep = 32;
    lb0 = (char*)lds_v[0] + c * 16;
    lb1 = (char*)lds_v[1] + c * 16;
  }
  async_b128(g, lb0);
  g += gstep;

  for (int it = 0; it < NPOS / 32; ++it) {
    WAIT_ASYNC0();
    __syncthreads();
    if (it + 1 < NPOS / 32) {
      async_b128(g, (it & 1) ? lb0 : lb1);    // next block into the other buffer
      g += gstep;
    }
    const _Float16* K = lds_k[it & 1];
    const _Float16* V = lds_v[it & 1];

    v16h kf0 = load_b_frag(K, 0);
    v16h kf1 = load_b_frag(K, 16);
    v8f z = {};
    v8f s0 = wmma_f16(qa.h, kf0, z);
    v8f s1 = wmma_f16(qa.h, kf1, z);

    float p0v[8], p1v[8];
#pragma unroll
    for (int r = 0; r < 8; ++r) {
      float a = s0[r] * ATT_SCALE_LOG2, cc = s1[r] * ATT_SCALE_LOG2;  // log2 domain
      float rm = fmaxf(a, cc);
      rm = fmaxf(rm, __shfl_xor(rm, 1));
      rm = fmaxf(rm, __shfl_xor(rm, 2));
      rm = fmaxf(rm, __shfl_xor(rm, 4));
      rm = fmaxf(rm, __shfl_xor(rm, 8));   // reduce within 16-lane half-group (same M)
      float mnew = fmaxf(m_i[r], rm);
      float alpha = __builtin_amdgcn_exp2f(m_i[r] - mnew);
      float e0 = __builtin_amdgcn_exp2f(a - mnew);
      float e1 = __builtin_amdgcn_exp2f(cc - mnew);
      float rs = e0 + e1;
      rs += __shfl_xor(rs, 1);
      rs += __shfl_xor(rs, 2);
      rs += __shfl_xor(rs, 4);
      rs += __shfl_xor(rs, 8);
      l_i[r] = l_i[r] * alpha + rs;
      m_i[r] = mnew;
      acc0[r] *= alpha;
      acc1[r] *= alpha;
      p0v[r] = e0; p1v[r] = e1;
    }

    // D-layout -> A-layout via per-wave LDS round trip (wave-private, no barrier)
    _Float16* P = lds_p[wv];
#pragma unroll
    for (int r = 0; r < 8; ++r) {
      int m = hf * 8 + r;
      P[m * 32 + n16]      = (_Float16)p0v[r];
      P[m * 32 + 16 + n16] = (_Float16)p1v[r];
    }
    v16h pf  = load_a_frag(P, 0);
    v16h vf0 = load_b_frag(V, 0);
    v16h vf1 = load_b_frag(V, 16);
    acc0 = wmma_f16(pf, vf0, acc0);
    acc1 = wmma_f16(pf, vf1, acc1);
  }

#pragma unroll
  for (int r = 0; r < 8; ++r) {
    float inv = 1.0f / l_i[r];
    int i = i0 + hf * 8 + r;
    size_t rowb = ((size_t)b * NPOS + i) * ADIM + h * HD;
    oT[rowb + n16]      = (_Float16)(acc0[r] * inv);
    oT[rowb + 16 + n16] = (_Float16)(acc1[r] * inv);
  }
}

// ---------------- Kernel 3: output projection Y = Wo * attn_out (f32 out) -----------
// Double-buffered: A staged via vectorized cvt path, B via async B128 copies.
__global__ __launch_bounds__(256) void cdna5_out_proj(
    const float* __restrict__ Wo, const _Float16* __restrict__ oT,
    float* __restrict__ Y) {
  const int b = blockIdx.z;
  const int o0 = blockIdx.y * 64;
  const int p0 = blockIdx.x * 128;
  __shared__ __align__(16) _Float16 lds_a[2][64 * 32];
  __shared__ __align__(16) _Float16 lds_b[2][128 * 32];
  const int t = threadIdx.x, lane = t & 31, wv = t >> 5;
  const int wr = wv >> 2, wc = wv & 3;
  v8f acc[2][2] = {};

  const int am = t >> 2, ac = (t & 3) * 8;
  const float* wp = Wo + (size_t)(o0 + am) * ADIM + ac;
  // async B: thread t stages rows n=t>>2 and n=(t>>2)+64, 16B each
  const _Float16* gB = oT + ((size_t)b * NPOS + p0 + (t >> 2)) * ADIM + (t & 3) * 8;

  auto stageA = [&](int nb) {
    float4 a0 = *(const float4*)(wp);
    float4 a1 = *(const float4*)(wp + 4);
    cvt8_store(&lds_a[nb][am * 32 + ac], a0, a1);
    wp += 32;
  };
  auto stageB = [&](int nb) {
    async_b128(gB, (char*)lds_b[nb] + t * 16);
    async_b128(gB + (size_t)64 * ADIM, (char*)lds_b[nb] + t * 16 + 4096);
    gB += 32;
  };

  stageA(0); stageB(0);
  for (int it = 0; it < 8; ++it) {
    WAIT_ASYNC0();
    __syncthreads();
    if (it + 1 < 8) { stageA((it + 1) & 1); stageB((it + 1) & 1); }
    const _Float16* la = lds_a[it & 1];
    const _Float16* lb = lds_b[it & 1];
    v16h af0 = load_a_frag(la, wr * 32);
    v16h af1 = load_a_frag(la, wr * 32 + 16);
    v16h bf0 = load_b_frag(lb, wc * 32);
    v16h bf1 = load_b_frag(lb, wc * 32 + 16);
    acc[0][0] = wmma_f16(af0, bf0, acc[0][0]);
    acc[0][1] = wmma_f16(af0, bf1, acc[0][1]);
    acc[1][0] = wmma_f16(af1, bf0, acc[1][0]);
    acc[1][1] = wmma_f16(af1, bf1, acc[1][1]);
  }

  const int hf = lane >> 4, n16 = lane & 15;
#pragma unroll
  for (int ti = 0; ti < 2; ++ti)
#pragma unroll
    for (int tj = 0; tj < 2; ++tj) {
      int p = p0 + wc * 32 + tj * 16 + n16;
#pragma unroll
      for (int r = 0; r < 8; ++r) {
        int o = o0 + wr * 32 + ti * 16 + hf * 8 + r;
        Y[((size_t)b * DIMC + o) * NPOS + p] = acc[ti][tj][r];
      }
    }
}

extern "C" void kernel_launch(void* const* d_in, const int* in_sizes, int n_in,
                              void* d_out, int out_size, void* d_ws, size_t ws_size,
                              hipStream_t stream) {
  (void)in_sizes; (void)n_in; (void)out_size; (void)ws_size;
  const float* query   = (const float*)d_in[0];
  const float* context = (const float*)d_in[1];
  const float* Wq = (const float*)d_in[2];
  const float* Wk = (const float*)d_in[3];
  const float* Wv = (const float*)d_in[4];
  const float* Wo = (const float*)d_in[5];
  float* Y = (float*)d_out;

  char* ws = (char*)d_ws;
  const size_t tsz = (size_t)4 * NHEAD * NPOS * HD * sizeof(_Float16);  // 8 MB
  _Float16* qT = (_Float16*)(ws);
  _Float16* kT = (_Float16*)(ws + tsz);
  _Float16* vS = (_Float16*)(ws + 2 * tsz);
  _Float16* oT = (_Float16*)(ws + 3 * tsz);

  cdna5_proj_qkv  <<<dim3(32, 4, 12), 256, 0, stream>>>(query, context, Wq, Wk, Wv,
                                                        qT, kT, vS);
  cdna5_flash_attn<<<dim3(32, 32, 1), 256, 0, stream>>>(qT, kT, vS, oT);
  cdna5_out_proj  <<<dim3(32, 8, 4),  256, 0, stream>>>(Wo, oT, Y);
}
